// SlimPhasorBlock_45337674776744
// MI455X (gfx1250) — matrix operand order repaired and checked by reading, not compiled, not run on hardware
//
#include <hip/hip_runtime.h>
#include <hip/hip_bf16.h>
#include <math.h>

// ---------------------------------------------------------------------------
// SlimPhasorBlock for MI455X (gfx1250): bf16 WMMA GEMMs with async
// Global->LDS B-tile staging (ASYNCcnt path) + coalesced scan kernels.
// B=4, L=4096, D=1024, P=128, V=8, M=B*L=16384.
// ---------------------------------------------------------------------------

typedef __bf16 bf16_t;
typedef __attribute__((ext_vector_type(16))) __bf16 bf16x16;
typedef __attribute__((ext_vector_type(8)))  __bf16 bf16x8;
typedef __attribute__((ext_vector_type(8)))  float  f32x8;
typedef __attribute__((ext_vector_type(4)))  int    v4i_t;

#define BB 4
#define LL 4096
#define DD 1024
#define PP 128
#define VV 8
#define MM (BB * LL)     // 16384
#define D2 (2 * DD)      // 2048
#define D3 (3 * DD)      // 3072

#define PI_F 3.14159265358979323846f

// ---- CDNA5 async Global->LDS copy (ASYNCcnt) with safe fallback -----------
// Probe-learned signature: (v4i AS1* src, v4i AS3* dst, Imm offset, Imm cpol)
#define AS1 __attribute__((address_space(1)))
#define AS3 __attribute__((address_space(3)))
#if defined(__gfx1250__) && __has_builtin(__builtin_amdgcn_global_load_async_to_lds_b128)
#define ASYNC_CP_B128(src, dst)                                          \
    __builtin_amdgcn_global_load_async_to_lds_b128(                      \
        (AS1 v4i_t*)(src), (AS3 v4i_t*)(dst), 0, 0)
#if __has_builtin(__builtin_amdgcn_s_wait_asynccnt)
#define WAIT_ASYNC(n) __builtin_amdgcn_s_wait_asynccnt(n)
#else
#define WAIT_ASYNC(n) asm volatile("s_wait_asynccnt %0" ::"n"(n) : "memory")
#endif
#else
#define ASYNC_CP_B128(src, dst) (*(bf16x8*)(dst) = *(const bf16x8*)(src))
#define WAIT_ASYNC(n) ((void)0)
#endif

// LDS B tile: 64 rows (N) x 32 (K) bf16, row pitch 40 elems (80 B, 16B-aligned;
// rows 0..15 start on 16 distinct bank groups since 20*n mod 64 is a bijection).
#define BP 40

// ---------------------------------------------------------------------------
// Generic WMMA bf16 GEMM:  C[M,N] = act(A[M,K] @ W[K,N] + bias) (+resid)
// A row-major bf16, Wt is W transposed: Wt[N][K] row-major bf16.
// Block = 256 threads = 8 waves; block tile 128x64; wave tile 16x64.
// B tile staged in LDS once per block (8x traffic cut), double-buffered via
// GLOBAL_LOAD_ASYNC_TO_LDS_B128 + s_wait_asynccnt. The last K-step is peeled
// so the steady-state loop body is branch-free (no per-iteration tail check).
// Fragment layouts per CDNA5 ISA 7.12.2:
//   A 16x32 bf16: lane = m + 16h (m=0..15), elems 0..7  = K[k0+8h .. +7],
//                                           elems 8..15 = K[k0+16+8h .. +7]
//   B 32x16 bf16: lane = n + 16h (n=0..15), elems 0..7  = K[k0+8h .. +7],
//                                           elems 8..15 = K[k0+16+8h .. +7]
//   C 16x16 f32 : lane = n + 16h, vgpr r -> row m = r + 8h, col n.
// act: 0 none, 1 tanh*pi, 2 sigmoid*|scale|, 3 exact gelu
// ---------------------------------------------------------------------------
__global__ __launch_bounds__(256) void gemm_wmma_bf16(
    const bf16_t* __restrict__ A, const bf16_t* __restrict__ Wt,
    const float* __restrict__ bias, const float* __restrict__ scale_ptr,
    const float* __restrict__ resid, float* __restrict__ C,
    int M, int N, int K, int act)
{
    __shared__ __align__(16) bf16_t ldsB[2][64 * BP];

    const int tid  = threadIdx.x;
    const int lane = tid & 31;
    const int wave = tid >> 5;
    const int nlan = lane & 15;
    const int hl   = lane >> 4;

    const int row0 = blockIdx.x * 128 + wave * 16;
    const int col0 = blockIdx.y * 64;

    // B-tile cooperative load mapping: thread t -> row ln (0..63), quarter lq.
    const int ln = tid >> 2;
    const int lq = tid & 3;
    const bf16_t* __restrict__ bsrc = Wt + (size_t)(col0 + ln) * K + lq * 8;
    const int bofs = ln * BP + lq * 8;

    f32x8 acc[4] = {};
    union BFrag { bf16x16 v; bf16x8 h[2]; };

    const bf16_t* __restrict__ arow = A + (size_t)(row0 + nlan) * K;

    // One K-step of fragment loads + 4 WMMAs from LDS buffer `buf`.
    auto compute_step = [&](int k0, int buf) {
        // CDNA5 prefetch path for the A stream (near caches).
        __builtin_prefetch(arow + k0 + 128, 0, 3);

        BFrag a;
        a.h[0] = *(const bf16x8*)(arow + k0 + hl * 8);
        a.h[1] = *(const bf16x8*)(arow + k0 + 16 + hl * 8);

#pragma unroll
        for (int j = 0; j < 4; ++j) {
            const bf16_t* bl = &ldsB[buf][(j * 16 + nlan) * BP + hl * 8];
            BFrag bfr;
            bfr.h[0] = *(const bf16x8*)(bl);
            bfr.h[1] = *(const bf16x8*)(bl + 16);
            acc[j] = __builtin_amdgcn_wmma_f32_16x16x32_bf16(
                false, a.v, false, bfr.v, (short)0, acc[j], false, false);
        }
    };

    // Prologue: stage first B tile into buffer 0.
    ASYNC_CP_B128(bsrc, &ldsB[0][bofs]);

    // Steady state: branch-free issue(next) -> wait(1) -> barrier -> compute.
    int k0 = 0;
    for (; k0 < K - 32; k0 += 32) {
        const int buf = (k0 >> 5) & 1;
        ASYNC_CP_B128(bsrc + k0 + 32, &ldsB[buf ^ 1][bofs]);
        WAIT_ASYNC(1);
        __syncthreads();
        compute_step(k0, buf);
        __syncthreads();  // readers done before next iteration overwrites buf
    }

    // Peeled last step: drain all outstanding async copies.
    WAIT_ASYNC(0);
    __syncthreads();
    compute_step(k0, (k0 >> 5) & 1);

    float sc = 1.0f;
    if (act == 2 && scale_ptr) sc = fabsf(scale_ptr[0]);

#pragma unroll
    for (int j = 0; j < 4; ++j) {
        const int col = col0 + j * 16 + nlan;
        const float bv = bias ? bias[col] : 0.0f;
#pragma unroll
        for (int r = 0; r < 8; ++r) {
            const int row = row0 + r + 8 * hl;
            float v = acc[j][r] + bv;
            if (act == 1)      v = tanhf(v) * PI_F;
            else if (act == 2) v = sc / (1.0f + __expf(-v));
            else if (act == 3) v = 0.5f * v * (1.0f + erff(v * 0.70710678118654752f));
            const size_t o = (size_t)row * N + col;
            if (resid) v += resid[o];
            C[o] = v;
        }
    }
}

// ------------------------- elementwise / staging ---------------------------

__global__ void transpose_w_k(const float* __restrict__ W, bf16_t* __restrict__ Wt,
                              int K, int N)
{
    size_t i = (size_t)blockIdx.x * blockDim.x + threadIdx.x;
    if (i >= (size_t)K * N) return;
    int n = (int)(i / K);
    int k = (int)(i % K);
    Wt[i] = (bf16_t)W[(size_t)k * N + n];
}

__global__ void f32_to_bf16_k(const float* __restrict__ in, bf16_t* __restrict__ out,
                              size_t n)
{
    size_t i = (size_t)blockIdx.x * blockDim.x + threadIdx.x;
    if (i < n) out[i] = (bf16_t)in[i];
}

// xb = bf16(x); off_in = [bf16(x) | bf16(pos)]; sk_in first half = bf16(x)
__global__ void build_inputs_k(const float* __restrict__ x, const float* __restrict__ pos,
                               bf16_t* __restrict__ xb, bf16_t* __restrict__ off_in,
                               bf16_t* __restrict__ sk_in)
{
    size_t i = (size_t)blockIdx.x * blockDim.x + threadIdx.x;
    if (i >= (size_t)MM * DD) return;
    size_t row = i / DD;
    int d = (int)(i % DD);
    int l = (int)(row % LL);
    float xv = x[i];
    bf16_t xh = (bf16_t)xv;
    xb[i] = xh;
    off_in[row * D2 + d] = xh;
    off_in[row * D2 + DD + d] = (bf16_t)pos[(size_t)l * DD + d];
    sk_in[row * D2 + d] = xh;
}

// ctx_avg scan: one thread per (b, d), coalesced in d.
__global__ void scan_ctx_k(const float* __restrict__ x, bf16_t* __restrict__ sk_in)
{
    int t = blockIdx.x * blockDim.x + threadIdx.x;
    if (t >= BB * DD) return;
    int b = t / DD, d = t % DD;
    float run = 0.0f;
    for (int l = 0; l < LL; ++l) {
        size_t row = (size_t)b * LL + l;
        run += x[row * DD + d];
        sk_in[row * D2 + DD + d] = (bf16_t)(run / (float)(l + 1));
    }
}

// mem1 phasor scan + rotation fusion -> bf16 A-matrix for pos_out GEMM.
__global__ void scan_mem1_k(const float* __restrict__ v1, const float* __restrict__ mag,
                            const float* __restrict__ offv, const float* __restrict__ pos,
                            bf16_t* __restrict__ mbf)
{
    int t = blockIdx.x * blockDim.x + threadIdx.x;
    if (t >= BB * DD) return;
    int b = t / DD, d = t % DD;
    const float rsD = rsqrtf((float)DD);
    float rm = 0.0f, rc = 0.0f, rs = 0.0f;
    for (int l = 0; l < LL; ++l) {
        size_t idx = ((size_t)b * LL + l) * DD + d;
        float ph = pos[(size_t)l * DD + d];
        float pc = cosf(ph), ps = sinf(ph);
        float m = mag[idx];
        float wv = m * v1[idx];
        rm += m; rc += pc * wv; rs += ps * wv;
        float ism = rsqrtf(rm + 1e-8f);
        float m1c = rc * ism, m1s = rs * ism;
        float o = offv[idx];
        float oc = cosf(o), osn = sinf(o);
        float qc = pc * oc - ps * osn;
        float qs = ps * oc + pc * osn;
        mbf[idx] = (bf16_t)((m1c * qc + m1s * qs) * rsD);
    }
}

// vals (V=8) + gate per row: one wave per row, lane-strided K reduction.
__global__ __launch_bounds__(256) void vals_gate_k(
    const float* __restrict__ x, const float* __restrict__ w_val,
    const float* __restrict__ b_val, const float* __restrict__ w_gate,
    const float* __restrict__ b_gate, float* __restrict__ vals,
    float* __restrict__ gate)
{
    int wave = (int)((blockIdx.x * (size_t)blockDim.x + threadIdx.x) >> 5);
    int lane = threadIdx.x & 31;
    if (wave >= MM) return;
    size_t row = (size_t)wave;
    float acc[VV] = {};
    float ag = 0.0f;
    for (int k = lane; k < DD; k += 32) {
        float xv = x[row * DD + k];
#pragma unroll
        for (int v = 0; v < VV; ++v) acc[v] += xv * w_val[(size_t)k * VV + v];
        ag += xv * w_gate[k];
    }
    for (int o = 16; o > 0; o >>= 1) {
        ag += __shfl_down(ag, o, 32);
#pragma unroll
        for (int v = 0; v < VV; ++v) acc[v] += __shfl_down(acc[v], o, 32);
    }
    if (lane == 0) {
#pragma unroll
        for (int v = 0; v < VV; ++v) vals[row * VV + v] = acc[v] + b_val[v];
        float g = ag + b_gate[0];
        gate[row] = 1.0f / (1.0f + __expf(-g));
    }
}

// kv phasor scan: one thread per (b, p, v); pv contiguous -> coalesced stores.
__global__ void scan_kv_k(const float* __restrict__ gate, const float* __restrict__ sp,
                          const float* __restrict__ vals, float* __restrict__ kvc,
                          float* __restrict__ kvs)
{
    int t = blockIdx.x * blockDim.x + threadIdx.x;
    if (t >= BB * PP * VV) return;
    int b = t / (PP * VV);
    int pv = t % (PP * VV);
    int p = pv / VV, v = pv % VV;
    float c = 0.0f, s = 0.0f, cg = 0.0f;
    for (int l = 0; l < LL; ++l) {
        size_t row = (size_t)b * LL + l;
        float g = gate[row];
        float a = sp[row * PP + p];
        float val = vals[row * VV + v] * g;
        c += cosf(a) * val;
        s += sinf(a) * val;
        cg += g;
        float ign = rsqrtf(fmaxf(cg, 1.0f));
        size_t o = row * (size_t)(PP * VV) + pv;
        kvc[o] = c * ign;
        kvs[o] = s * ign;
    }
}

// kv retrieval + qpc mean: one wave per row, reduce over P.
__global__ __launch_bounds__(256) void kv_reduce_k(
    const float* __restrict__ qp, const float* __restrict__ kvc,
    const float* __restrict__ kvs, float* __restrict__ kv_ret,
    float* __restrict__ qmean)
{
    int wave = (int)((blockIdx.x * (size_t)blockDim.x + threadIdx.x) >> 5);
    int lane = threadIdx.x & 31;
    if (wave >= MM) return;
    size_t row = (size_t)wave;
    float acc[VV] = {};
    float qcm = 0.0f;
    for (int p = lane; p < PP; p += 32) {
        float q = qp[row * PP + p];
        float qc = cosf(q), qs = sinf(q);
        qcm += qc;
        size_t base = row * (size_t)(PP * VV) + (size_t)p * VV;
#pragma unroll
        for (int v = 0; v < VV; ++v) acc[v] += qc * kvc[base + v] + qs * kvs[base + v];
    }
    for (int o = 16; o > 0; o >>= 1) {
        qcm += __shfl_down(qcm, o, 32);
#pragma unroll
        for (int v = 0; v < VV; ++v) acc[v] += __shfl_down(acc[v], o, 32);
    }
    if (lane == 0) {
        const float rsP = rsqrtf((float)PP);
#pragma unroll
        for (int v = 0; v < VV; ++v) kv_ret[row * VV + v] = acc[v] * rsP;
        qmean[row] = qcm / (float)PP;
    }
}

// kv_out = kv_ret @ w_kv + b_kv (K = V = 8: plain VALU, coalesced in d).
__global__ void kv_out_k(const float* __restrict__ kv_ret, const float* __restrict__ w_kv,
                         const float* __restrict__ b_kv, float* __restrict__ kv_out)
{
    size_t i = (size_t)blockIdx.x * blockDim.x + threadIdx.x;
    if (i >= (size_t)MM * DD) return;
    size_t row = i / DD;
    int d = (int)(i % DD);
    float acc = b_kv[d];
#pragma unroll
    for (int v = 0; v < VV; ++v) acc += kv_ret[row * VV + v] * w_kv[(size_t)v * DD + d];
    kv_out[i] = acc;
}

__device__ __forceinline__ float block_reduce_sum(float v, float* red)
{
    int tid = threadIdx.x;
    red[tid] = v;
    __syncthreads();
    for (int s = 128; s > 0; s >>= 1) {
        if (tid < s) red[tid] += red[tid + s];
        __syncthreads();
    }
    float r = red[0];
    __syncthreads();
    return r;
}

// comb = [pos_out | kv_out | x*qmean]; LayerNorm over 3072; bf16 out.
__global__ __launch_bounds__(256) void comb_norm_k(
    const float* __restrict__ pos_out, const float* __restrict__ kv_out,
    const float* __restrict__ x, const float* __restrict__ qmean,
    const float* __restrict__ ln_g, const float* __restrict__ ln_b,
    bf16_t* __restrict__ comb_n)
{
    __shared__ float sh[D3];
    __shared__ float red[256];
    size_t row = blockIdx.x;
    float qm = qmean[row];
    float lsum = 0.0f, lsq = 0.0f;
    for (int j = threadIdx.x; j < D3; j += 256) {
        float v;
        if (j < DD)          v = pos_out[row * DD + j];
        else if (j < 2 * DD) v = kv_out[row * DD + (j - DD)];
        else                 v = x[row * DD + (j - 2 * DD)] * qm;
        sh[j] = v;
        lsum += v;
        lsq += v * v;
    }
    __syncthreads();
    float tot = block_reduce_sum(lsum, red);
    float tot2 = block_reduce_sum(lsq, red);
    float mu = tot / (float)D3;
    float var = tot2 / (float)D3 - mu * mu;
    float rstd = rsqrtf(var + 1e-5f);
    for (int j = threadIdx.x; j < D3; j += 256) {
        float v = (sh[j] - mu) * rstd * ln_g[j] + ln_b[j];
        comb_n[row * D3 + j] = (bf16_t)v;
    }
}

// ---------------------------------------------------------------------------

extern "C" void kernel_launch(void* const* d_in, const int* in_sizes, int n_in,
                              void* d_out, int out_size, void* d_ws, size_t ws_size,
                              hipStream_t stream)
{
    (void)in_sizes; (void)n_in; (void)out_size; (void)ws_size;

    const float* x      = (const float*)d_in[0];
    const float* pos    = (const float*)d_in[1];
    const float* w_mem1v = (const float*)d_in[2];
    const float* b_mem1v = (const float*)d_in[3];
    const float* w_mem1o = (const float*)d_in[4];
    const float* b_mem1o = (const float*)d_in[5];
    const float* w_off   = (const float*)d_in[6];
    const float* b_off   = (const float*)d_in[7];
    const float* w_mag   = (const float*)d_in[8];
    const float* b_mag   = (const float*)d_in[9];
    const float* mag_scale = (const float*)d_in[10];
    const float* w_key   = (const float*)d_in[11];
    const float* b_key   = (const float*)d_in[12];
    const float* w_val   = (const float*)d_in[13];
    const float* b_val   = (const float*)d_in[14];
    const float* w_sk1   = (const float*)d_in[15];
    const float* b_sk1   = (const float*)d_in[16];
    const float* w_sk2   = (const float*)d_in[17];
    const float* b_sk2   = (const float*)d_in[18];
    const float* w_gate  = (const float*)d_in[19];
    const float* b_gate  = (const float*)d_in[20];
    const float* w_kv    = (const float*)d_in[21];
    const float* b_kv    = (const float*)d_in[22];
    const float* ln_g    = (const float*)d_in[23];
    const float* ln_b    = (const float*)d_in[24];
    const float* w_o1    = (const float*)d_in[25];
    const float* b_o1    = (const float*)d_in[26];
    const float* w_o2    = (const float*)d_in[27];
    const float* b_o2    = (const float*)d_in[28];

    float* outp = (float*)d_out;

    // ---------------- workspace carve-out ----------------
    char* base = (char*)d_ws;
    size_t cur = 0;
    auto alloc = [&](size_t bytes) -> void* {
        void* p = base + cur;
        cur = (cur + bytes + 255) & ~(size_t)255;
        return p;
    };

    // bf16 transposed weights (Wt[N][K])
    bf16_t* wt_mem1v = (bf16_t*)alloc((size_t)DD * DD * 2);
    bf16_t* wt_mem1o = (bf16_t*)alloc((size_t)DD * DD * 2);
    bf16_t* wt_off   = (bf16_t*)alloc((size_t)DD * D2 * 2);
    bf16_t* wt_mag   = (bf16_t*)alloc((size_t)DD * DD * 2);
    bf16_t* wt_key   = (bf16_t*)alloc((size_t)PP * DD * 2);
    bf16_t* wt_sk1   = (bf16_t*)alloc((size_t)DD * D2 * 2);
    bf16_t* wt_sk2   = (bf16_t*)alloc((size_t)PP * DD * 2);
    bf16_t* wt_o1    = (bf16_t*)alloc((size_t)D2 * D3 * 2);
    bf16_t* wt_o2    = (bf16_t*)alloc((size_t)DD * D2 * 2);

    // bf16 activations
    bf16_t* xb       = (bf16_t*)alloc((size_t)MM * DD * 2);
    bf16_t* off_in_b = (bf16_t*)alloc((size_t)MM * D2 * 2);
    bf16_t* sk_in_b  = (bf16_t*)alloc((size_t)MM * D2 * 2);
    bf16_t* mbf      = (bf16_t*)alloc((size_t)MM * DD * 2);
    bf16_t* comb_n   = (bf16_t*)alloc((size_t)MM * D3 * 2);

    // fp32 intermediates
    float* v1      = (float*)alloc((size_t)MM * DD * 4);
    float* mag     = (float*)alloc((size_t)MM * DD * 4);
    float* offv    = (float*)alloc((size_t)MM * DD * 4);
    float* qp      = (float*)alloc((size_t)MM * PP * 4);
    float* sp      = (float*)alloc((size_t)MM * PP * 4);
    float* vals    = (float*)alloc((size_t)MM * VV * 4);
    float* gate    = (float*)alloc((size_t)MM * 4);
    float* sk1h    = (float*)alloc((size_t)MM * DD * 4);
    float* pos_out = (float*)alloc((size_t)MM * DD * 4);
    float* kv_ret  = (float*)alloc((size_t)MM * VV * 4);
    float* qmean   = (float*)alloc((size_t)MM * 4);
    float* h1      = (float*)alloc((size_t)MM * D2 * 4);

    // lifetime-based reuse (P*V == D so sizes match)
    float*  kvc    = v1;      // free after scan_mem1
    float*  kvs    = mag;     // free after scan_mem1
    float*  kv_out = offv;    // free after scan_mem1
    bf16_t* sk1h_b = xb;      // free after v1/mag/key GEMMs
    bf16_t* h1b    = sk_in_b; // free after sk1 GEMM

    auto grid1 = [](size_t n) { return (unsigned)((n + 255) / 256); };

    auto gemm = [&](const bf16_t* A, const bf16_t* Wt, const float* bias,
                    const float* scp, const float* resid, float* Cm,
                    int Mm, int Nn, int Kk, int act) {
        dim3 g(Mm / 128, Nn / 64);
        gemm_wmma_bf16<<<g, 256, 0, stream>>>(A, Wt, bias, scp, resid, Cm,
                                              Mm, Nn, Kk, act);
    };

    // 1) weights -> bf16 transposed
    transpose_w_k<<<grid1((size_t)DD * DD), 256, 0, stream>>>(w_mem1v, wt_mem1v, DD, DD);
    transpose_w_k<<<grid1((size_t)DD * DD), 256, 0, stream>>>(w_mem1o, wt_mem1o, DD, DD);
    transpose_w_k<<<grid1((size_t)DD * D2), 256, 0, stream>>>(w_off,   wt_off,   D2, DD);
    transpose_w_k<<<grid1((size_t)DD * DD), 256, 0, stream>>>(w_mag,   wt_mag,   DD, DD);
    transpose_w_k<<<grid1((size_t)PP * DD), 256, 0, stream>>>(w_key,   wt_key,   DD, PP);
    transpose_w_k<<<grid1((size_t)DD * D2), 256, 0, stream>>>(w_sk1,   wt_sk1,   D2, DD);
    transpose_w_k<<<grid1((size_t)PP * DD), 256, 0, stream>>>(w_sk2,   wt_sk2,   DD, PP);
    transpose_w_k<<<grid1((size_t)D2 * D3), 256, 0, stream>>>(w_o1,    wt_o1,    D3, D2);
    transpose_w_k<<<grid1((size_t)DD * D2), 256, 0, stream>>>(w_o2,    wt_o2,    D2, DD);

    // 2) staging + ctx scan
    build_inputs_k<<<grid1((size_t)MM * DD), 256, 0, stream>>>(x, pos, xb, off_in_b, sk_in_b);
    scan_ctx_k<<<grid1(BB * DD), 256, 0, stream>>>(x, sk_in_b);

    // 3) front GEMMs
    gemm(xb,       wt_mem1v, b_mem1v, nullptr,   nullptr, v1,   MM, DD, DD, 0);
    gemm(xb,       wt_mag,   b_mag,   mag_scale, nullptr, mag,  MM, DD, DD, 2);
    gemm(off_in_b, wt_off,   b_off,   nullptr,   nullptr, offv, MM, DD, D2, 1);
    gemm(xb,       wt_key,   b_key,   nullptr,   nullptr, qp,   MM, PP, DD, 1);
    vals_gate_k<<<MM / 8, 256, 0, stream>>>(x, w_val, b_val, w_gate, b_gate, vals, gate);

    // 4) skim path
    gemm(sk_in_b, wt_sk1, b_sk1, nullptr, nullptr, sk1h, MM, DD, D2, 3);
    f32_to_bf16_k<<<grid1((size_t)MM * DD), 256, 0, stream>>>(sk1h, sk1h_b, (size_t)MM * DD);
    gemm(sk1h_b, wt_sk2, b_sk2, nullptr, nullptr, sp, MM, PP, DD, 1);

    // 5) mem1 phasor scan + output projection
    scan_mem1_k<<<grid1(BB * DD), 256, 0, stream>>>(v1, mag, offv, pos, mbf);
    gemm(mbf, wt_mem1o, b_mem1o, nullptr, nullptr, pos_out, MM, DD, DD, 0);

    // 6) kv phasor scan + retrieval
    scan_kv_k<<<grid1(BB * PP * VV), 256, 0, stream>>>(gate, sp, vals, kvc, kvs);
    kv_reduce_k<<<MM / 8, 256, 0, stream>>>(qp, kvc, kvs, kv_ret, qmean);
    kv_out_k<<<grid1((size_t)MM * DD), 256, 0, stream>>>(kv_ret, w_kv, b_kv, kv_out);

    // 7) concat + LayerNorm -> bf16
    comb_norm_k<<<MM, 256, 0, stream>>>(pos_out, kv_out, x, qmean, ln_g, ln_b, comb_n);

    // 8) output MLP (o2 fuses residual add of x)
    gemm(comb_n, wt_o1, b_o1, nullptr, nullptr, h1, MM, D2, D3, 3);
    f32_to_bf16_k<<<grid1((size_t)MM * D2), 256, 0, stream>>>(h1, h1b, (size_t)MM * D2);
    gemm(h1b, wt_o2, b_o2, nullptr, x, outp, MM, DD, D2, 0);
}
